// SSA_38225208934979
// MI455X (gfx1250) — matrix operand order repaired and checked by reading, not compiled, not run on hardware
//
#include <hip/hip_runtime.h>

typedef _Float16 v16h __attribute__((ext_vector_type(16)));
typedef _Float16 v8h  __attribute__((ext_vector_type(8)));
typedef _Float16 v4h  __attribute__((ext_vector_type(4)));
typedef float    v8f  __attribute__((ext_vector_type(8)));
typedef float    v4f  __attribute__((ext_vector_type(4)));
typedef unsigned u32x4 __attribute__((ext_vector_type(4)));
typedef int      i32x4 __attribute__((ext_vector_type(4)));
typedef int      i32x8 __attribute__((ext_vector_type(8)));

// Problem constants
#define TOKENS 16384   // B*S
#define SEQ    4096
#define NHEAD  16

#if __has_builtin(__builtin_amdgcn_tensor_load_to_lds) && __has_builtin(__builtin_amdgcn_s_wait_tensorcnt)
#define HAVE_TDM 1
#else
#define HAVE_TDM 0
#endif

__device__ __forceinline__ float siluf(float x) { return x * (1.0f / (1.0f + __expf(-x))); }
// order-preserving float<->uint encoding for atomicMax on possibly-negative floats
__device__ __forceinline__ unsigned fenc(float f) {
  unsigned u = __float_as_uint(f);
  return (u & 0x80000000u) ? ~u : (u | 0x80000000u);
}
__device__ __forceinline__ float fdec(unsigned k) {
  return (k & 0x80000000u) ? __uint_as_float(k ^ 0x80000000u) : __uint_as_float(~k);
}
__device__ __forceinline__ v16h cat8(v8h lo, v8h hi) {
  return __builtin_shufflevector(lo, hi, 0,1,2,3,4,5,6,7,8,9,10,11,12,13,14,15);
}
// A fragment (16x32 f16): lane<16 -> row=lane, K {0..7, 16..23}; lane>=16 -> K {8..15, 24..31}
__device__ __forceinline__ v16h load_afrag(const _Float16* row_k0, int lh) {
  v8h lo = *(const v8h*)(row_k0 + 8 * lh);
  v8h hi = *(const v8h*)(row_k0 + 16 + 8 * lh);
  return cat8(lo, hi);
}
// B fragment (32x16 f16): lane<16 -> col=lane, K {0..7, 16..23}; lane>=16 -> K {8..15, 24..31}
__device__ __forceinline__ v16h load_bfrag(const _Float16* col_k0, int lh) {
  const _Float16* p = col_k0 + 8 * lh;
  v8h lo = *(const v8h*)(p);
  v8h hi = *(const v8h*)(p + 16);
  return cat8(lo, hi);
}
__device__ __forceinline__ v8f wmma_f16(v16h a, v16h b, v8f c) {
  return __builtin_amdgcn_wmma_f32_16x16x32_f16(false, a, false, b, (short)0, c, false, false);
}

#if HAVE_TDM
// TDM: DMA a 128-row x 32-half tile (row stride = strideK halves in memory) into LDS,
// padded 4 DWORDs after every 16 DWORDs -> LDS row stride 40 halves (bank-conflict-free).
__device__ __forceinline__ void tdm_load_tile_128x32(const _Float16* gsrc, void* lds_dst,
                                                     int strideK) {
  unsigned lds = (unsigned)(unsigned long long)lds_dst;             // LDS offset = addr[31:0]
  unsigned long long ga = (unsigned long long)gsrc;
  u32x4 g0;
  g0[0] = 1u;                                                        // count=1, user descriptor
  g0[1] = lds;                                                       // lds_addr
  g0[2] = (unsigned)ga;                                              // global_addr[31:0]
  g0[3] = ((unsigned)(ga >> 32) & 0x01FFFFFFu) | (2u << 30);         // addr[56:32] | type=2
  i32x8 g1;
  g1[0] = (int)((1u << 16)      // data_size = 2 bytes
              | (1u << 20)      // pad_enable
              | (3u << 22)      // pad_interval: 16 DWORDs
              | (3u << 25));    // pad_amount: 4 DWORDs
  g1[1] = (int)(32u << 16);                    // tensor_dim0[15:0] (=tile: full tiles only)
  g1[2] = (int)(128u << 16);                   // tensor_dim0 hi | tensor_dim1[15:0]=128
  g1[3] = (int)(32u << 16);                    // tensor_dim1 hi | tile_dim0=32
  g1[4] = (int)128u;                           // tile_dim1=128 (tile_dim2=0)
  g1[5] = (int)(unsigned)strideK;              // tensor_dim0_stride (data_size units)
  g1[6] = 0;
  g1[7] = 0;
  i32x4 z4 = {};
#if __clang_major__ >= 23
  i32x8 z8 = {};
  __builtin_amdgcn_tensor_load_to_lds(g0, g1, z4, z4, z8, 0);
#else
  __builtin_amdgcn_tensor_load_to_lds(g0, g1, z4, z4, 0);
#endif
}
#endif

// ---------------------------------------------------------------------------
// Generic tiled WMMA GEMM:  C = A(MxK) @ W(NxK)^T   (A f32 or f16, W f16)
// Block = 256 threads (8 waves), tile 128x128, BK=32, per-wave 32x64 (2x4 frags).
// f16-A path (full tiles): double-buffered TDM tensor_load_to_lds for A & W tiles.
// f32-A path: manual cvt+stage (handles ragged N for wkv_a).
// MODE: 0 plain f16 out | 1 kv_a split | 2 q proj (silu+eta / raw pe)
//       3 kv_b (silu k + eta / v transposed) | 4 f32 out
// ---------------------------------------------------------------------------
template <int AF32, int MODE>
__global__ __launch_bounds__(256) void gemm_wmma(
    const void* __restrict__ Avoid, const _Float16* __restrict__ W,
    int M, int N, int K,
    _Float16* __restrict__ O16, _Float16* __restrict__ O16b,
    float* __restrict__ Of, unsigned* __restrict__ etaRaw, int S)
{
  __shared__ _Float16 sA[2][128][40];   // stride 40 halves = 80B (16B aligned, conflict-free)
  __shared__ _Float16 sW[2][128][40];
  __shared__ float wred[8];

  const int tid = threadIdx.x;
  const int lane = tid & 31, wave = tid >> 5;
  const int lh = lane >> 4, lr = lane & 15;
  const int waveM = wave & 3, waveN = wave >> 2;
  const int row0 = blockIdx.y * 128, col0 = blockIdx.x * 128;

  const float* Af = (const float*)Avoid;
  const _Float16* Ah = (const _Float16*)Avoid;

  v8f acc[2][4] = {};

  for (int k0 = 0; k0 < K; k0 += 32) {
    int p = 0;
#if HAVE_TDM
    if (!AF32) {
      p = (k0 >> 5) & 1;
      if (k0 == 0 && wave == 0) {       // prologue: DMA first tiles
        tdm_load_tile_128x32(Ah + (size_t)row0 * K, &sA[0][0][0], K);
        tdm_load_tile_128x32(W + (size_t)col0 * K, &sW[0][0][0], K);
      }
      if (wave == 0) __builtin_amdgcn_s_wait_tensorcnt(0);
      __syncthreads();                  // buffer p visible to all; buffer 1-p reads retired
      if (wave == 0 && k0 + 32 < K) {   // overlap next DMA with this step's WMMAs
        tdm_load_tile_128x32(Ah + (size_t)row0 * K + k0 + 32, &sA[1 - p][0][0], K);
        tdm_load_tile_128x32(W + (size_t)col0 * K + k0 + 32, &sW[1 - p][0][0], K);
      }
    } else
#endif
    {
      __syncthreads();
      if (AF32) {
        for (int i = 0; i < 4; i++) {           // 128x32 f32 -> cvt f16
          int q = tid + i * 256;
          int r = q >> 3, c4 = q & 7;
          v4f x = *(const v4f*)(Af + (size_t)(row0 + r) * K + k0 + c4 * 4);
          v4h hx;
          hx[0] = (_Float16)x[0]; hx[1] = (_Float16)x[1];
          hx[2] = (_Float16)x[2]; hx[3] = (_Float16)x[3];
          *(v4h*)&sA[0][r][c4 * 4] = hx;
        }
      } else {
        for (int i = 0; i < 2; i++) {           // 128x32 f16 (TDM-less fallback)
          int q = tid + i * 256;
          int r = q >> 2, c = q & 3;
          *(v8h*)&sA[0][r][c * 8] = *(const v8h*)(Ah + (size_t)(row0 + r) * K + k0 + c * 8);
        }
      }
      for (int i = 0; i < 2; i++) {             // W tile 128x32 f16 (row = out col)
        int q = tid + i * 256;
        int r = q >> 2, c = q & 3;
        int wr = col0 + r;
        v8h wv = {};
        if (wr < N) wv = *(const v8h*)(W + (size_t)wr * K + k0 + c * 8);
        *(v8h*)&sW[0][r][c * 8] = wv;
      }
      {  // speculative prefetch of next K-slab (emits global_prefetch_b8)
        int pr = col0 + (tid & 127);
        if (k0 + 64 <= K - 32 && pr < N) __builtin_prefetch(W + (size_t)pr * K + k0 + 64, 0, 1);
      }
      __syncthreads();
    }

    v16h a[2], b[4];
    for (int mf = 0; mf < 2; mf++) a[mf] = load_afrag(&sA[p][waveM * 32 + mf * 16 + lr][0], lh);
    for (int nf = 0; nf < 4; nf++) b[nf] = load_bfrag(&sW[p][waveN * 64 + nf * 16 + lr][0], lh);
    for (int mf = 0; mf < 2; mf++)
      for (int nf = 0; nf < 4; nf++)
        acc[mf][nf] = wmma_f16(a[mf], b[nf], acc[mf][nf]);
  }

  // ---------------- epilogue ----------------
  float lmax = -3.0e38f;
  for (int mf = 0; mf < 2; mf++) {
    for (int nf = 0; nf < 4; nf++) {
      int col = col0 + waveN * 64 + nf * 16 + lr;
      for (int g = 0; g < 8; g++) {
        int row = row0 + waveM * 32 + mf * 16 + g + 8 * lh;
        float v = acc[mf][nf][g];
        if (MODE == 0) {
          O16[(size_t)row * N + col] = (_Float16)v;
        } else if (MODE == 1) {
          if (col < N) {
            if (col < 512) O16[(size_t)row * 512 + col] = (_Float16)v;
            else           O16b[(size_t)row * 64 + (col - 512)] = (_Float16)v;
          }
        } else if (MODE == 2) {
          int h = col / 192, r = col % 192;
          if (r < 128) {
            float sv = siluf(v); lmax = fmaxf(lmax, sv);
            O16[(size_t)row * 2048 + h * 128 + r] = (_Float16)sv;
          } else {
            O16b[(size_t)row * 1024 + h * 64 + (r - 128)] = (_Float16)v;
          }
        } else if (MODE == 3) {
          int h = col / 256, r = col % 256;
          if (r < 128) {
            float sv = siluf(v); lmax = fmaxf(lmax, sv);
            O16[(size_t)row * 2048 + h * 128 + r] = (_Float16)sv;
          } else {
            int d = r - 128, bb = row / S, sr = row % S;
            O16b[(((size_t)bb * NHEAD + h) * 128 + d) * S + sr] = (_Float16)v;  // vT (b,h,dv,s)
          }
        } else {
          Of[(size_t)row * N + col] = v;
        }
      }
    }
  }
  if (MODE == 2 || MODE == 3) {  // fused eta chunk-max (tile lies in one 256-token chunk)
    for (int off = 16; off > 0; off >>= 1) lmax = fmaxf(lmax, __shfl_xor(lmax, off, 32));
    if (lane == 0) wred[wave] = lmax;
    __syncthreads();
    if (tid == 0) {
      float m = wred[0];
      for (int i = 1; i < 8; i++) m = fmaxf(m, wred[i]);
      int bb = row0 / S, ch = (row0 % S) >> 8;
      atomicMax(etaRaw + bb * NHEAD + ch, fenc(m));
    }
  }
}

// ---------------------------------------------------------------------------
// RoPE + SiLU for q_pe (16384 tokens x 16 heads x 64) + eta_qp chunk max
// ---------------------------------------------------------------------------
__global__ __launch_bounds__(256) void rope_silu_q(
    const _Float16* __restrict__ praw, const float* __restrict__ freqs,
    _Float16* __restrict__ qp, unsigned* __restrict__ etaRaw)
{
  __shared__ float wred[8];
  int tid = threadIdx.x, lane = tid & 31, wave = tid >> 5;
  int tok0 = blockIdx.x * 16;
  float lmax = -3.0e38f;
  for (int j = 0; j < 32; j++) {
    int p = tid + j * 256;                 // 16 tok * 16 head * 32 pairs = 8192
    int tl = p >> 9, rem = p & 511, h = rem >> 5, i = rem & 31;
    int tok = tok0 + tl, s = tok & (SEQ - 1);
    float c = freqs[(s * 32 + i) * 2], sn = freqs[(s * 32 + i) * 2 + 1];
    size_t base = (size_t)tok * 1024 + h * 64 + 2 * i;
    float xr = (float)praw[base], xi = (float)praw[base + 1];
    float sr = siluf(xr * c - xi * sn), si = siluf(xr * sn + xi * c);
    qp[base] = (_Float16)sr; qp[base + 1] = (_Float16)si;
    lmax = fmaxf(lmax, fmaxf(sr, si));
  }
  for (int off = 16; off > 0; off >>= 1) lmax = fmaxf(lmax, __shfl_xor(lmax, off, 32));
  if (lane == 0) wred[wave] = lmax;
  __syncthreads();
  if (tid == 0) {
    float m = wred[0];
    for (int i = 1; i < 8; i++) m = fmaxf(m, wred[i]);
    atomicMax(etaRaw + (tok0 >> 12) * NHEAD + ((tok0 & (SEQ - 1)) >> 8), fenc(m));
  }
}

// RoPE + SiLU for k_pe (16384 tokens x 64) + eta_kp chunk max
__global__ __launch_bounds__(256) void rope_silu_k(
    const _Float16* __restrict__ praw, const float* __restrict__ freqs,
    _Float16* __restrict__ kpo, unsigned* __restrict__ etaRaw)
{
  __shared__ float wred[8];
  int tid = threadIdx.x, lane = tid & 31, wave = tid >> 5;
  int tok0 = blockIdx.x * 64;
  float lmax = -3.0e38f;
  for (int j = 0; j < 8; j++) {
    int p = tid + j * 256;                 // 64 tok * 32 pairs = 2048
    int tl = p >> 5, i = p & 31;
    int tok = tok0 + tl, s = tok & (SEQ - 1);
    float c = freqs[(s * 32 + i) * 2], sn = freqs[(s * 32 + i) * 2 + 1];
    size_t base = (size_t)tok * 64 + 2 * i;
    float xr = (float)praw[base], xi = (float)praw[base + 1];
    float sr = siluf(xr * c - xi * sn), si = siluf(xr * sn + xi * c);
    kpo[base] = (_Float16)sr; kpo[base + 1] = (_Float16)si;
    lmax = fmaxf(lmax, fmaxf(sr, si));
  }
  for (int off = 16; off > 0; off >>= 1) lmax = fmaxf(lmax, __shfl_xor(lmax, off, 32));
  if (lane == 0) wred[wave] = lmax;
  __syncthreads();
  if (tid == 0) {
    float m = wred[0];
    for (int i = 1; i < 8; i++) m = fmaxf(m, wred[i]);
    atomicMax(etaRaw + (tok0 >> 12) * NHEAD + ((tok0 & (SEQ - 1)) >> 8), fenc(m));
  }
}

__global__ void eta_init(unsigned* etaRaw) { etaRaw[threadIdx.x] = fenc(-3.0e38f); }
__global__ void eta_final(const unsigned* __restrict__ etaRaw, float* __restrict__ etaF) {
  int i = threadIdx.x;
  float m = fdec(etaRaw[i]);
  if (m == 0.0f) m = 1e-6f;
  etaF[i] = fminf(10.0f / m, 1.0f);
}

__global__ __launch_bounds__(256) void cvt_f32_f16(const float* __restrict__ in,
                                                   _Float16* __restrict__ out, int n) {
  int i = blockIdx.x * 256 + threadIdx.x;
  if (i < n) out[i] = (_Float16)in[i];
}

// ---------------------------------------------------------------------------
// Block-local causal attention, one (b, block, head) per 128-thread block.
// scores = cn * (qn @ kn^T) + cp * (qp @ kp^T); softmax; out = attn @ v.
// ---------------------------------------------------------------------------
__global__ __launch_bounds__(128) void ssa_attn(
    const _Float16* __restrict__ qn, const _Float16* __restrict__ kn,
    const _Float16* __restrict__ qp, const _Float16* __restrict__ kp,
    const _Float16* __restrict__ vT, const float* __restrict__ etaF,
    _Float16* __restrict__ out)
{
  __shared__ _Float16 attnS[64][72];
  const int tid = threadIdx.x;
  const int lane = tid & 31, wave = tid >> 5;
  const int lh = lane >> 4, lr = lane & 15;
  const int h = blockIdx.x, n = blockIdx.y, b = blockIdx.z;
  const int t0 = b * SEQ + n * 64;   // global token base
  const int s0 = n * 64;             // within-batch base
  const float SCALE = 0.07216878364870323f;  // (128+64)^-0.5
  const float cn = etaF[0   + b * 16 + h] * etaF[64  + b * 16 + h] * SCALE;
  const float cp = etaF[128 + b * 16 + h] * etaF[192 + b * 16 + h] * SCALE;

  v8f accN[4] = {}, accP[4] = {};
  const int qrow = t0 + wave * 16 + lr;
  for (int kk = 0; kk < 4; kk++) {                       // K=128 nope
    v16h a = load_afrag(qn + (size_t)qrow * 2048 + h * 128 + kk * 32, lh);
    for (int nf = 0; nf < 4; nf++) {
      v16h bf = load_bfrag(kn + (size_t)(t0 + nf * 16 + lr) * 2048 + h * 128 + kk * 32, lh);
      accN[nf] = wmma_f16(a, bf, accN[nf]);
    }
  }
  for (int kk = 0; kk < 2; kk++) {                       // K=64 pe
    v16h a = load_afrag(qp + (size_t)qrow * 1024 + h * 64 + kk * 32, lh);
    for (int nf = 0; nf < 4; nf++) {
      v16h bf = load_bfrag(kp + (size_t)(t0 + nf * 16 + lr) * 64 + kk * 32, lh);
      accP[nf] = wmma_f16(a, bf, accP[nf]);
    }
  }
  // combine + causal mask
  float sc[4][8];
  for (int nf = 0; nf < 4; nf++)
    for (int g = 0; g < 8; g++) {
      int qi = wave * 16 + g + 8 * lh;
      int kj = nf * 16 + lr;
      float v = accN[nf][g] * cn + accP[nf][g] * cp;
      sc[nf][g] = (kj > qi) ? -3.0e38f : v;
    }
  // row softmax (row spread across 16 lanes x 4 N-frags)
  for (int g = 0; g < 8; g++) {
    float m = fmaxf(fmaxf(sc[0][g], sc[1][g]), fmaxf(sc[2][g], sc[3][g]));
    for (int off = 8; off > 0; off >>= 1) m = fmaxf(m, __shfl_xor(m, off, 16));
    float s = 0.0f;
    for (int nf = 0; nf < 4; nf++) { sc[nf][g] = __expf(sc[nf][g] - m); s += sc[nf][g]; }
    for (int off = 8; off > 0; off >>= 1) s += __shfl_xor(s, off, 16);
    float inv = 1.0f / s;
    for (int nf = 0; nf < 4; nf++) sc[nf][g] *= inv;
  }
  for (int nf = 0; nf < 4; nf++)
    for (int g = 0; g < 8; g++)
      attnS[wave * 16 + g + 8 * lh][nf * 16 + lr] = (_Float16)sc[nf][g];
  __syncthreads();

  // out = attn (64x64) @ v (64x128); v stored transposed (b,h,dv,s) -> contiguous B frags
  v8f accO[8] = {};
  for (int kk = 0; kk < 2; kk++) {
    v16h a = load_afrag(&attnS[wave * 16 + lr][kk * 32], lh);
    for (int df = 0; df < 8; df++) {
      const _Float16* vp = vT + ((size_t)(b * NHEAD + h) * 128 + df * 16 + lr) * SEQ + s0 + kk * 32;
      v16h bf = load_bfrag(vp, lh);
      accO[df] = wmma_f16(a, bf, accO[df]);
    }
  }
  for (int df = 0; df < 8; df++)
    for (int g = 0; g < 8; g++) {
      int row = t0 + wave * 16 + g + 8 * lh;
      out[(size_t)row * 2048 + h * 128 + df * 16 + lr] = (_Float16)accO[df][g];
    }
}

// ---------------------------------------------------------------------------
extern "C" void kernel_launch(void* const* d_in, const int* in_sizes, int n_in,
                              void* d_out, int out_size, void* d_ws, size_t ws_size,
                              hipStream_t stream)
{
  (void)in_sizes; (void)n_in; (void)out_size; (void)ws_size;
  const float* x     = (const float*)d_in[0];
  const float* freqs = (const float*)d_in[1];
  const float* wqa   = (const float*)d_in[2];
  const float* wqb   = (const float*)d_in[3];
  const float* wkva  = (const float*)d_in[4];
  const float* wkvb  = (const float*)d_in[5];
  const float* wo    = (const float*)d_in[6];
  float* out = (float*)d_out;

  char* p = (char*)d_ws;
  auto take = [&](size_t bytes) { char* r = p; p += bytes; return r; };

  _Float16* wqa16   = (_Float16*)take((size_t)512 * 2048 * 2);
  _Float16* wqb16   = (_Float16*)take((size_t)3072 * 512 * 2);
  _Float16* wkva16  = (_Float16*)take((size_t)576 * 2048 * 2);
  _Float16* wkvb16  = (_Float16*)take((size_t)4096 * 512 * 2);
  _Float16* wo16    = (_Float16*)take((size_t)2048 * 2048 * 2);
  _Float16* qlat16  = (_Float16*)take((size_t)TOKENS * 512 * 2);   // dead after G3
  _Float16* kvc16   = (_Float16*)take((size_t)TOKENS * 512 * 2);   // dead after G4
  _Float16* qperaw  = (_Float16*)take((size_t)TOKENS * 1024 * 2);  // dead after rope_q
  _Float16* qn16    = (_Float16*)take((size_t)TOKENS * 2048 * 2);
  _Float16* qp16    = (_Float16*)take((size_t)TOKENS * 1024 * 2);
  _Float16* k16     = (_Float16*)take((size_t)TOKENS * 2048 * 2);
  _Float16* vT16    = (_Float16*)take((size_t)TOKENS * 2048 * 2);  // (b,h,dv,s)
  _Float16* kp16    = (_Float16*)take((size_t)TOKENS * 64 * 2);
  _Float16* kperaw  = (_Float16*)take((size_t)TOKENS * 64 * 2);
  unsigned* etaRaw  = (unsigned*)take(256 * sizeof(unsigned));     // qn|kn|qp|kp, 64 each
  float*    etaF    = (float*)take(256 * sizeof(float));
  // attn_out aliases the dead (qlat|kvc|qperaw) region: exactly 64 MiB
  _Float16* attn16  = qlat16;

  // 1) weights -> f16
  cvt_f32_f16<<<(512 * 2048 + 255) / 256, 256, 0, stream>>>(wqa, wqa16, 512 * 2048);
  cvt_f32_f16<<<(3072 * 512 + 255) / 256, 256, 0, stream>>>(wqb, wqb16, 3072 * 512);
  cvt_f32_f16<<<(576 * 2048 + 255) / 256, 256, 0, stream>>>(wkva, wkva16, 576 * 2048);
  cvt_f32_f16<<<(4096 * 512 + 255) / 256, 256, 0, stream>>>(wkvb, wkvb16, 4096 * 512);
  cvt_f32_f16<<<(2048 * 2048 + 255) / 256, 256, 0, stream>>>(wo, wo16, 2048 * 2048);
  eta_init<<<1, 256, 0, stream>>>(etaRaw);

  const int MT = TOKENS / 128;  // 128 row tiles
  // 2) q_lat = x @ wq_a^T
  gemm_wmma<1, 0><<<dim3(512 / 128, MT), 256, 0, stream>>>(
      x, wqa16, TOKENS, 512, 2048, qlat16, nullptr, nullptr, nullptr, SEQ);
  // 3) kv = x @ wkv_a^T  (split kv_c | k_pe_raw)
  gemm_wmma<1, 1><<<dim3((576 + 127) / 128, MT), 256, 0, stream>>>(
      x, wkva16, TOKENS, 576, 2048, kvc16, kperaw, nullptr, nullptr, SEQ);
  // 4) q = q_lat @ wq_b^T  (silu nope -> qn + eta_qn, raw pe -> qperaw)  [TDM path]
  gemm_wmma<0, 2><<<dim3(3072 / 128, MT), 256, 0, stream>>>(
      qlat16, wqb16, TOKENS, 3072, 512, qn16, qperaw, nullptr, etaRaw + 0, SEQ);
  // 5) kv = kv_c @ wkv_b^T  (silu k -> k16 + eta_kn, v -> vT)            [TDM path]
  gemm_wmma<0, 3><<<dim3(4096 / 128, MT), 256, 0, stream>>>(
      kvc16, wkvb16, TOKENS, 4096, 512, k16, vT16, nullptr, etaRaw + 64, SEQ);
  // 6) rope+silu pe paths + eta_qp / eta_kp
  rope_silu_q<<<TOKENS / 16, 256, 0, stream>>>(qperaw, freqs, qp16, etaRaw + 128);
  rope_silu_k<<<TOKENS / 64, 256, 0, stream>>>(kperaw, freqs, kp16, etaRaw + 192);
  eta_final<<<1, 256, 0, stream>>>(etaRaw, etaF);
  // 7) block-causal attention -> attn_out f16
  ssa_attn<<<dim3(NHEAD, SEQ / 64, 4), 128, 0, stream>>>(
      qn16, k16, qp16, kp16, vT16, etaF, attn16);
  // 8) final projection: out = attn_out @ wo^T  (f32 to d_out)           [TDM path]
  gemm_wmma<0, 4><<<dim3(2048 / 128, MT), 256, 0, stream>>>(
      attn16, wo16, TOKENS, 2048, 2048, nullptr, nullptr, out, nullptr, SEQ);
}
